// MoERouter_28381143892385
// MI455X (gfx1250) — compile-verified, matrix operand order checked by reading, not run on hardware
//
#include <hip/hip_runtime.h>

#define N_TOKENS    32768
#define HIDDEN      2048
#define NUM_EXPERTS 64
#define KT_COUNT    (HIDDEN / 32)   // 64 k-tiles of 32
#define ROWPAD      67              // LDS row stride (floats), odd to avoid bank conflicts
#define WAVES_PER_BLOCK 8
#define TOKENS_PER_BLOCK 256

typedef __bf16 v16bf __attribute__((ext_vector_type(16)));
typedef float  v8f   __attribute__((ext_vector_type(8)));
typedef float  f32x4 __attribute__((ext_vector_type(4)));

// ---------------------------------------------------------------------------
// Kernel 1: swizzle gate_w [2048,64] f32 -> bf16 WMMA B-operand layout, and
// zero the global accumulators (probSum[64], cnt[64]).
// B layout per lane (mirror of ISA 16-bit A layout):
//   lane<16 : N=lane,    elems j<8 -> K=k0+j (k0=kt*32),   j>=8 -> K=k0+16+(j-8)
//   lane>=16: N=lane-16, k0 = kt*32 + 8
// Buffer layout: [kt][nt][lane][16 bf16]  (each lane entry = 32 contiguous bytes)
// ---------------------------------------------------------------------------
__global__ void moe_prep(const float* __restrict__ gate_w,
                         __bf16* __restrict__ Bsw,
                         float* __restrict__ gAcc) {
    const int tid = blockIdx.x * blockDim.x + threadIdx.x;
    if (tid < 2 * NUM_EXPERTS) gAcc[tid] = 0.0f;
    if (tid >= KT_COUNT * 4 * 32) return;

    const int lane = tid & 31;
    const int nt   = (tid >> 5) & 3;
    const int kt   = tid >> 7;

    const int n  = nt * 16 + (lane & 15);
    const int k0 = kt * 32 + ((lane >> 4) ? 8 : 0);
    const int k1 = k0 + 16;

    __bf16* dst = Bsw + (size_t)tid * 16;
#pragma unroll
    for (int j = 0; j < 8; ++j)
        dst[j] = (__bf16)gate_w[(k0 + j) * NUM_EXPERTS + n];
#pragma unroll
    for (int j = 0; j < 8; ++j)
        dst[8 + j] = (__bf16)gate_w[(k1 + j) * NUM_EXPERTS + n];
}

// ---------------------------------------------------------------------------
// Kernel 2: main router. 8 waves/block, 32 tokens/wave, 256 tokens/block.
// Per wave: logits[32x64] via v_wmma_f32_16x16x32_bf16, epilogue via LDS.
// ---------------------------------------------------------------------------
__global__ __launch_bounds__(256) void moe_router(const float* __restrict__ x,
                                                  const __bf16* __restrict__ Bsw,
                                                  float* __restrict__ gAcc,
                                                  float* __restrict__ out) {
    __shared__ float sLog[WAVES_PER_BLOCK][16 * ROWPAD];
    __shared__ float sProb[NUM_EXPERTS];
    __shared__ float sCnt[NUM_EXPERTS];

    const int lane = threadIdx.x & 31;
    const int wave = threadIdx.x >> 5;
    if (threadIdx.x < NUM_EXPERTS) {
        sProb[threadIdx.x] = 0.0f;
        sCnt[threadIdx.x]  = 0.0f;
    }
    __syncthreads();

    const int tokenBase = blockIdx.x * TOKENS_PER_BLOCK + wave * 32;
    const int hi  = lane >> 4;    // 0 or 1 (lane half)
    const int l16 = lane & 15;

    const float* rowA0 = x + (size_t)(tokenBase + l16) * HIDDEN;
    const float* rowA1 = x + (size_t)(tokenBase + 16 + l16) * HIDDEN;

    v8f acc[2][4] = {};

    for (int kt = 0; kt < KT_COUNT; ++kt) {
        const int ka0 = kt * 32 + hi * 8;
        const int ka1 = ka0 + 16;

        // Prefetch next k-tile of x (speculative; safe if it runs past the row)
        __builtin_prefetch(rowA0 + ka0 + 32, 0, 1);
        __builtin_prefetch(rowA1 + ka0 + 32, 0, 1);

        // Build bf16 A operands from f32 (two 8-float chunks per lane, per tile)
        v16bf A0, A1;
        {
            const f32x4* p0 = (const f32x4*)(rowA0 + ka0);
            const f32x4* p1 = (const f32x4*)(rowA0 + ka1);
            f32x4 u0 = p0[0], u1 = p0[1], u2 = p1[0], u3 = p1[1];
#pragma unroll
            for (int j = 0; j < 4; ++j) {
                A0[j]      = (__bf16)u0[j];
                A0[4 + j]  = (__bf16)u1[j];
                A0[8 + j]  = (__bf16)u2[j];
                A0[12 + j] = (__bf16)u3[j];
            }
        }
        {
            const f32x4* p0 = (const f32x4*)(rowA1 + ka0);
            const f32x4* p1 = (const f32x4*)(rowA1 + ka1);
            f32x4 u0 = p0[0], u1 = p0[1], u2 = p1[0], u3 = p1[1];
#pragma unroll
            for (int j = 0; j < 4; ++j) {
                A1[j]      = (__bf16)u0[j];
                A1[4 + j]  = (__bf16)u1[j];
                A1[8 + j]  = (__bf16)u2[j];
                A1[12 + j] = (__bf16)u3[j];
            }
        }

#pragma unroll
        for (int nt = 0; nt < 4; ++nt) {
            v16bf B = *(const v16bf*)(Bsw + (((size_t)kt * 4 + nt) * 32 + lane) * 16);
            acc[0][nt] = __builtin_amdgcn_wmma_f32_16x16x32_bf16(
                false, A0, false, B, (short)0, acc[0][nt], false, false);
            acc[1][nt] = __builtin_amdgcn_wmma_f32_16x16x32_bf16(
                false, A1, false, B, (short)0, acc[1][nt], false, false);
        }
    }

    // ---- Epilogue: one 16x64 tile at a time through this wave's LDS slab ----
    float* myLds = sLog[wave];
    for (int mt = 0; mt < 2; ++mt) {
        // C layout: VGPR r, lane L -> M = r + 8*(L>=16), N = (L&15) + 16*nt
#pragma unroll
        for (int nt = 0; nt < 4; ++nt)
#pragma unroll
            for (int r = 0; r < 8; ++r)
                myLds[(r + 8 * hi) * ROWPAD + nt * 16 + l16] = acc[mt][nt][r];

        if (lane < 16) {
            const int token = tokenBase + mt * 16 + lane;
            const float* row = myLds + lane * ROWPAD;

            float m1 = -3.4e38f, m2 = -3.4e38f;
            int i1 = 0, i2 = 0;
            for (int e = 0; e < NUM_EXPERTS; ++e) {
                float v = row[e];
                if (v > m1)      { m2 = m1; i2 = i1; m1 = v; i1 = e; }
                else if (v > m2) { m2 = v; i2 = e; }
            }
            float s = 0.0f;
            for (int e = 0; e < NUM_EXPERTS; ++e) s += __expf(row[e] - m1);
            const float inv = 1.0f / s;
            for (int e = 0; e < NUM_EXPERTS; ++e)
                atomicAdd(&sProb[e], __expf(row[e] - m1) * inv);

            atomicAdd(&sCnt[i1], 1.0f);
            atomicAdd(&sCnt[i2], 1.0f);

            const float w1 = 1.0f / (1.0f + __expf(m2 - m1));
            out[token * 2 + 0] = w1;
            out[token * 2 + 1] = 1.0f - w1;
            out[2 * N_TOKENS + token * 2 + 0] = (float)i1;
            out[2 * N_TOKENS + token * 2 + 1] = (float)i2;
        }
    }

    __syncthreads();
    if (threadIdx.x < NUM_EXPERTS) {
        atomicAdd(&gAcc[threadIdx.x], sProb[threadIdx.x]);               // probSum
        atomicAdd(&gAcc[NUM_EXPERTS + threadIdx.x], sCnt[threadIdx.x]);  // counts
    }
}

// ---------------------------------------------------------------------------
// Kernel 3: loss = E * sum_e (cnt_e/N) * (probSum_e/N)
// ---------------------------------------------------------------------------
__global__ void moe_loss(const float* __restrict__ gAcc, float* __restrict__ out) {
    __shared__ float red[NUM_EXPERTS];
    const int t = threadIdx.x;
    red[t] = gAcc[t] * gAcc[NUM_EXPERTS + t];
    __syncthreads();
    for (int s = NUM_EXPERTS / 2; s > 0; s >>= 1) {
        if (t < s) red[t] += red[t + s];
        __syncthreads();
    }
    if (t == 0)
        out[4 * N_TOKENS] =
            (float)NUM_EXPERTS * red[0] / ((float)N_TOKENS * (float)N_TOKENS);
}

extern "C" void kernel_launch(void* const* d_in, const int* in_sizes, int n_in,
                              void* d_out, int out_size, void* d_ws, size_t ws_size,
                              hipStream_t stream) {
    const float* x      = (const float*)d_in[0];
    const float* gate_w = (const float*)d_in[1];
    float* out = (float*)d_out;

    // Workspace: [ Bsw: 64*4*32*16 bf16 = 256KB ][ gAcc: 128 f32 ]
    __bf16* Bsw = (__bf16*)d_ws;
    float* gAcc = (float*)((char*)d_ws + (size_t)KT_COUNT * 4 * 32 * 16 * sizeof(__bf16));

    moe_prep<<<32, 256, 0, stream>>>(gate_w, Bsw, gAcc);
    moe_router<<<N_TOKENS / TOKENS_PER_BLOCK, 256, 0, stream>>>(x, Bsw, gAcc, out);
    moe_loss<<<1, NUM_EXPERTS, 0, stream>>>(gAcc, out);
}
